// LigerFusedLinearKTOLoss_46454366273587
// MI455X (gfx1250) — matrix operand order repaired and checked by reading, not compiled, not run on hardware
//
#include <hip/hip_runtime.h>
#include <hip/hip_bf16.h>
#include <stdint.h>

// ---------------------------------------------------------------------------
// Fused Linear KTO loss for MI455X (gfx1250, wave32, WMMA).
//
// Pipeline:
//   1) kto_cvt_f32_to_f16: W (V,H) f32 -> f16 once per matrix (into d_ws).
//   2) kto_lse_kernel: per 16-row tile, sweep V in 256-col chunks
//      (8 waves x two 16x16 WMMA tiles sharing one A fragment).
//      Logits via v_wmma_f32_16x16x32_f16 with f32 accumulate, explicit
//      double-buffered fragment loads (loads overlap WMMA), online
//      logsumexp + target-logit capture, per-block partial seq sums.
//   3) kto_final_kernel: B=4 KTO loss from partials (fixed-order sums).
//
// Workspace layout (requires ws_size >= 2*V*H*2 + 256*4 bytes ~= 131.1 MB):
//   [0, V*H*2)            : policy weight f16
//   [V*H*2, 2*V*H*2)      : ref weight f16
//   [2*V*H*2, +1KB)       : float part[2][128] per-block token-logp sums
// ---------------------------------------------------------------------------

#define KTO_H      1024
#define KTO_V      32000
#define KTO_BT     2048      // B*T rows
#define KTO_T      512
#define KTO_ROWS   16        // rows per block (one seq spans 32 blocks)
#define KTO_BCOLS  256       // columns per V-chunk (8 waves x 2 tiles x 16)
#define KTO_BETA   0.1f
#define KTO_IGNORE (-100)

typedef _Float16 v16h __attribute__((ext_vector_type(16)));
typedef _Float16 v8h  __attribute__((ext_vector_type(8)));
typedef _Float16 v4h  __attribute__((ext_vector_type(4)));
typedef float    v8f  __attribute__((ext_vector_type(8)));

// -------------------------------------------------------------- convert ----
__global__ __launch_bounds__(256) void kto_cvt_f32_to_f16(
    const float* __restrict__ src, _Float16* __restrict__ dst, int n) {
  int stride = gridDim.x * blockDim.x * 4;
  for (int i = (blockIdx.x * blockDim.x + threadIdx.x) * 4; i < n; i += stride) {
    const float4 v = *(const float4*)(src + i);
    v4h o;
    o[0] = (_Float16)v.x; o[1] = (_Float16)v.y;
    o[2] = (_Float16)v.z; o[3] = (_Float16)v.w;
    *(v4h*)(dst + i) = o;
  }
}

// ------------------------------------------------------ fused GEMM + LSE ----
__global__ __launch_bounds__(256) void kto_lse_kernel(
    const float* __restrict__ x0, const _Float16* __restrict__ w0,
    const float* __restrict__ b0,
    const float* __restrict__ x1, const _Float16* __restrict__ w1,
    const float* __restrict__ b1,
    const int* __restrict__ target,
    float* __restrict__ part) {

  const int mat = blockIdx.y;
  const float*    __restrict__ x  = mat ? x1 : x0;
  const _Float16* __restrict__ w  = mat ? w1 : w0;
  const float*    __restrict__ bi = mat ? b1 : b0;

  // LDS: activation tile in f16 (+8 halves pad per row), staged logits, scratch.
  __shared__ _Float16 sA[KTO_ROWS][KTO_H + 8];     // 33 KB
  __shared__ float sLog[KTO_ROWS][KTO_BCOLS + 4];  // 16.6 KB, stride 260
  __shared__ float sPM[KTO_ROWS][16];
  __shared__ float sPS[KTO_ROWS][16];
  __shared__ float sM[KTO_ROWS], sS[KTO_ROWS], sT[KTO_ROWS], sRed[KTO_ROWS];
  __shared__ int   sTgt[KTO_ROWS];

  const int tid = threadIdx.x;
  const int r0  = blockIdx.x * KTO_ROWS;

  // Stage x tile (16 x 1024) into LDS as f16, coalesced float4 reads.
  for (int i = tid; i < KTO_ROWS * (KTO_H / 4); i += 256) {
    const int row = i / (KTO_H / 4);
    const int k4  = (i % (KTO_H / 4)) * 4;
    const float4 v = *(const float4*)(x + (size_t)(r0 + row) * KTO_H + k4);
    sA[row][k4 + 0] = (_Float16)v.x;
    sA[row][k4 + 1] = (_Float16)v.y;
    sA[row][k4 + 2] = (_Float16)v.z;
    sA[row][k4 + 3] = (_Float16)v.w;
  }
  if (tid < KTO_ROWS) {
    sM[tid] = -1e30f; sS[tid] = 0.0f; sT[tid] = 0.0f;
    sTgt[tid] = target[r0 + tid];
  }
  __syncthreads();

  const int wave = tid >> 5;          // 0..7
  const int lane = tid & 31;
  const int n    = lane & 15;         // N index / A row (M)
  const int half = lane >> 4;         // lane half selects K sub-range

  // A fragment source row in LDS (same 16 rows for all waves; LDS broadcast).
  const _Float16* __restrict__ ap = &sA[n][half * 16];

  for (int it = 0; it < KTO_V / KTO_BCOLS; ++it) {
    const int cb   = it * KTO_BCOLS;
    const int col0 = cb + wave * 16 + n;          // tile 0 (cols cb..cb+127)
    const int col1 = col0 + 128;                  // tile 1 (cols cb+128..cb+255)
    const _Float16* __restrict__ wp0 = w + (size_t)col0 * KTO_H + half * 16;
    const _Float16* __restrict__ wp1 = w + (size_t)col1 * KTO_H + half * 16;
    const float bv0 = bi[col0];
    const float bv1 = bi[col1];

    // Double-buffered fragments: prime k=0, prefetch k+32 before each WMMA.
    v8h alo  = *(const v8h*)(ap);
    v8h ahi  = *(const v8h*)(ap + 8);
    v8h b0lo = *(const v8h*)(wp0);
    v8h b0hi = *(const v8h*)(wp0 + 8);
    v8h b1lo = *(const v8h*)(wp1);
    v8h b1hi = *(const v8h*)(wp1 + 8);

    v8f c0 = {}, c1 = {};
    #pragma unroll
    for (int k0 = 0; k0 < KTO_H; k0 += 32) {
      v16h a, bb0, bb1;
      #pragma unroll
      for (int q = 0; q < 8; ++q) {
        a[q]   = alo[q];  a[q + 8]   = ahi[q];
        bb0[q] = b0lo[q]; bb0[q + 8] = b0hi[q];
        bb1[q] = b1lo[q]; bb1[q + 8] = b1hi[q];
      }
      if (k0 + 32 < KTO_H) {   // compile-time constant under full unroll
        alo  = *(const v8h*)(ap  + k0 + 32);
        ahi  = *(const v8h*)(ap  + k0 + 40);
        b0lo = *(const v8h*)(wp0 + k0 + 32);
        b0hi = *(const v8h*)(wp0 + k0 + 40);
        b1lo = *(const v8h*)(wp1 + k0 + 32);
        b1hi = *(const v8h*)(wp1 + k0 + 40);
      }
      c0 = __builtin_amdgcn_wmma_f32_16x16x32_f16(
               false, a, false, bb0, (short)0, c0, false, false);
      c1 = __builtin_amdgcn_wmma_f32_16x16x32_f16(
               false, a, false, bb1, (short)0, c1, false, false);
    }

    // D layout: element j -> M = j + 8*half, N = n.
    #pragma unroll
    for (int j = 0; j < 8; ++j) {
      sLog[j + 8 * half][wave * 16 + n]       = c0[j] + bv0;
      sLog[j + 8 * half][128 + wave * 16 + n] = c1[j] + bv1;
    }
    __syncthreads();

    // Partial max / sum-exp: 256 threads = 16 rows x 16 slots x 16 cols.
    {
      const int row = tid >> 4, slot = tid & 15;
      float vv[16], mx = -1e30f;
      #pragma unroll
      for (int q = 0; q < 16; ++q) { vv[q] = sLog[row][slot * 16 + q]; mx = fmaxf(mx, vv[q]); }
      float s = 0.0f;
      #pragma unroll
      for (int q = 0; q < 16; ++q) s += __expf(vv[q] - mx);
      sPM[row][slot] = mx; sPS[row][slot] = s;
    }
    __syncthreads();

    // Online logsumexp merge + target logit capture (one thread per row).
    if (tid < KTO_ROWS) {
      const int row = tid;
      const float mOld = sM[row], sOld = sS[row];
      float mNew = mOld;
      #pragma unroll
      for (int q = 0; q < 16; ++q) mNew = fmaxf(mNew, sPM[row][q]);
      float s = sOld * __expf(mOld - mNew);
      #pragma unroll
      for (int q = 0; q < 16; ++q) s += sPS[row][q] * __expf(sPM[row][q] - mNew);
      sM[row] = mNew; sS[row] = s;
      const int tc = sTgt[row] - cb;
      if (tc >= 0 && tc < KTO_BCOLS) sT[row] = sLog[row][tc];
    }
    __syncthreads();
  }

  // Per-row target log-prob, then fixed-order block sum (deterministic).
  if (tid < KTO_ROWS) {
    const int row = tid;
    const float lse = sM[row] + __logf(sS[row]);
    sRed[row] = (sTgt[row] != KTO_IGNORE) ? (sT[row] - lse) : 0.0f;
  }
  __syncthreads();
  if (tid == 0) {
    float acc = 0.0f;
    #pragma unroll
    for (int r = 0; r < KTO_ROWS; ++r) acc += sRed[r];
    part[mat * 128 + blockIdx.x] = acc;   // 16-row tile lies inside one sequence
  }
}

// ------------------------------------------------------------- final loss ----
__global__ void kto_final_kernel(const float* __restrict__ part,
                                 const unsigned char* __restrict__ prefs,
                                 const float* __restrict__ kl,
                                 float* __restrict__ out) {
  if (threadIdx.x != 0 || blockIdx.x != 0) return;
  const float k0 = kl[0];
  float loss = 0.0f, ch = 0.0f, rj = 0.0f;
  for (int s = 0; s < 4; ++s) {
    float pol = 0.0f, ref = 0.0f;
    for (int b = 0; b < 32; ++b) {          // 512 rows / 16 rows-per-block
      pol += part[s * 32 + b];
      ref += part[128 + s * 32 + b];
    }
    const float lr   = pol - ref;
    const float mult = prefs[s] ? 1.0f : -1.0f;
    const float z    = KTO_BETA * (lr - k0) * mult;
    const float sig  = 1.0f / (1.0f + __expf(-z));
    loss += 1.0f - sig;
    const float rw = KTO_BETA * lr;
    if (prefs[s]) ch += rw; else rj += rw;
  }
  out[0] = loss / (float)KTO_BT;
  out[1] = ch;
  out[2] = rj;
}

// ---------------------------------------------------------------- launch ----
extern "C" void kernel_launch(void* const* d_in, const int* in_sizes, int n_in,
                              void* d_out, int out_size, void* d_ws, size_t ws_size,
                              hipStream_t stream) {
  const float* x0 = (const float*)d_in[0];          // _input  (2048,1024)
  const float* w0 = (const float*)d_in[1];          // weight  (32000,1024)
  const int*   tg = (const int*)  d_in[2];          // target  (2048,)
  const float* b0 = (const float*)d_in[3];          // bias    (32000,)
  const unsigned char* pf = (const unsigned char*)d_in[4]; // preference_labels (4,)
  const float* x1 = (const float*)d_in[5];          // ref_input
  const float* w1 = (const float*)d_in[6];          // ref_weight
  const float* b1 = (const float*)d_in[7];          // ref_bias
  const float* kl = (const float*)d_in[8];          // kl (1,)

  const size_t wn = (size_t)KTO_V * KTO_H;          // 32,768,000 elements
  _Float16* w16_0 = (_Float16*)d_ws;
  _Float16* w16_1 = w16_0 + wn;
  float* part = (float*)((char*)d_ws + 2 * wn * sizeof(_Float16));

  kto_cvt_f32_to_f16<<<2048, 256, 0, stream>>>(w0, w16_0, (int)wn);
  kto_cvt_f32_to_f16<<<2048, 256, 0, stream>>>(w1, w16_1, (int)wn);

  dim3 grid(KTO_BT / KTO_ROWS, 2);                  // 128 row tiles x {pol, ref}
  kto_lse_kernel<<<grid, 256, 0, stream>>>(x0, w16_0, b0, x1, w16_1, b1, tg, part);

  kto_final_kernel<<<1, 32, 0, stream>>>(part, pf, kl, (float*)d_out);
}
